// SPNet_22393959482101
// MI455X (gfx1250) — matrix operand order repaired and checked by reading, not compiled, not run on hardware
//
#include <hip/hip_runtime.h>
#include <hip/hip_bf16.h>
#include <hip/hip_fp16.h>

typedef __attribute__((ext_vector_type(16))) _Float16 v16h;
typedef __attribute__((ext_vector_type(8)))  _Float16 v8h;
typedef __attribute__((ext_vector_type(8)))  float    v8f;
typedef __attribute__((ext_vector_type(4)))  float    v4f;

static constexpr int VOX  = 200000;  // V in reference
static constexpr int NCLS = 20;      // NUM_CLASSES

__device__ inline v8f wmma32(v16h a, v16h b, v8f c) {
  // D = A(16x32 f16) * B(32x16 f16) + C(16x16 f32)
  return __builtin_amdgcn_wmma_f32_16x16x32_f16(false, a, false, b, (short)0, c, false, false);
}

#define CAT16(lo, hi) __builtin_shufflevector((lo), (hi), 0, 1, 2, 3, 4, 5, 6, 7, \
                                              8, 9, 10, 11, 12, 13, 14, 15)

// ---- fragment builders (ISA 7.12.2 layouts, wave32) -----------------------
// A 16xK tile from row-major f16 buffer (16 rows, leading dim ld in {32,64},
// K-chunk k0 in {0,32}); all chunk addresses are 16B aligned -> b128 loads.
__device__ inline v16h load_a_f16(const _Float16* src, int ld, int k0) {
  const int lane = threadIdx.x & 31;
  const int row  = lane & 15;
  const int kb   = (lane >> 4) * 8;       // 0 for lanes 0-15, 8 for 16-31
  const _Float16* p = src + row * ld + k0 + kb;
  v8h lo = *(const v8h*)(p);              // K = k0 + kb + 0..7
  v8h hi = *(const v8h*)(p + 16);         // K = k0 + 16 + kb + 0..7
  return CAT16(lo, hi);
}
// B 32x16 tile from row-major f16 weights w[K][ld], K-chunk k0, col tile n0.
// (strided by ld -> scalar loads; built once per wave, amortized)
__device__ inline v16h load_b_f16(const _Float16* w, int ld, int k0, int n0) {
  const int lane = threadIdx.x & 31;
  const int col  = n0 + (lane & 15);
  const int kb   = (lane >> 4) * 8;       // lanes 0-15: K 0..7,16..23; 16-31: K 8..15,24..31
  v16h b;
#pragma unroll
  for (int i = 0; i < 16; ++i) b[i] = w[(size_t)(k0 + kb + (i < 8 ? i : i + 8)) * ld + col];
  return b;
}

__device__ inline void lds_fence() { asm volatile("s_wait_dscnt 0" ::: "memory"); }

// ===========================================================================
// Kernel 1: point MLP 4->32->64->32 (BN folded, ReLU), write pf (f16) and
//           segment_max scatter into vox via int atomicMax (values >= 0).
// ===========================================================================
__global__ __launch_bounds__(256) void k_point_encode(
    const float* __restrict__ points, const int* __restrict__ pv,
    const float* __restrict__ w1, const float* __restrict__ g1, const float* __restrict__ b1,
    const float* __restrict__ w2, const float* __restrict__ g2, const float* __restrict__ b2,
    const float* __restrict__ w3, const float* __restrict__ g3, const float* __restrict__ b3,
    _Float16* __restrict__ pf16, float* __restrict__ vox, int N) {
  __shared__ float s_w1[4 * 32];
  __shared__ float s_b1[32];
  __shared__ alignas(16) _Float16 s_w2[32 * 64];
  __shared__ float s_b2[64];
  __shared__ alignas(16) _Float16 s_w3[64 * 32];
  __shared__ float s_b3[32];
  __shared__ alignas(16) _Float16 s_h2[8][16 * 64];   // per-wave 16x64 tile

  const int tid = threadIdx.x;
  const float inv = rsqrtf(1.0f + 1e-5f);
  for (int i = tid; i < 4 * 32; i += 256)  s_w1[i] = w1[i] * (g1[i & 31] * inv);
  for (int i = tid; i < 32;     i += 256)  s_b1[i] = b1[i];
  for (int i = tid; i < 32 * 64; i += 256) s_w2[i] = (_Float16)(w2[i] * (g2[i & 63] * inv));
  for (int i = tid; i < 64;     i += 256)  s_b2[i] = b2[i];
  for (int i = tid; i < 64 * 32; i += 256) s_w3[i] = (_Float16)(w3[i] * (g3[i & 31] * inv));
  for (int i = tid; i < 32;     i += 256)  s_b3[i] = b3[i];
  __syncthreads();

  const int wave = tid >> 5, lane = tid & 31;
  const int n  = lane & 15;
  const int mb = (lane >> 4) * 8;
  const int kb = (lane >> 4) * 8;

  v16h b2f[4];
#pragma unroll
  for (int t = 0; t < 4; ++t) b2f[t] = load_b_f16(s_w2, 64, 0, t * 16);
  v16h b3f[2][2];
#pragma unroll
  for (int t = 0; t < 2; ++t) {
    b3f[t][0] = load_b_f16(s_w3, 32, 0,  t * 16);
    b3f[t][1] = load_b_f16(s_w3, 32, 32, t * 16);
  }

  const int ntiles = (N + 15) >> 4;
  const int nwaves = gridDim.x * 8;
  for (int tile = blockIdx.x * 8 + wave; tile < ntiles; tile += nwaves) {
    const int row0 = tile << 4;
    const int myrow = row0 + n;
    const int r = myrow < N ? myrow : N - 1;
    if (myrow + 16 < N) __builtin_prefetch(&points[(size_t)(myrow + 16) * 5], 0, 1);

    // layer 1 (K=4, scalar) directly into A-fragment layout for layer 2
    const float x0 = points[(size_t)r * 5 + 1], x1 = points[(size_t)r * 5 + 2];
    const float x2 = points[(size_t)r * 5 + 3], x3 = points[(size_t)r * 5 + 4];
    v16h a1;
#pragma unroll
    for (int h = 0; h < 16; ++h) {
      int k = (h < 8) ? (kb + h) : (16 + kb + (h - 8));
      float v = fmaf(x0, s_w1[k], fmaf(x1, s_w1[32 + k],
                fmaf(x2, s_w1[64 + k], fmaf(x3, s_w1[96 + k], s_b1[k]))));
      a1[h] = (_Float16)fmaxf(v, 0.0f);
    }

    // layer 2: 16x32 @ 32x64 -> 16x64, bias+ReLU, spill f16 tile to LDS
    _Float16* h2 = s_h2[wave];
#pragma unroll
    for (int t = 0; t < 4; ++t) {
      v8f c = {};
      c = wmma32(a1, b2f[t], c);
      const int col = t * 16 + n;
      const float bias = s_b2[col];
#pragma unroll
      for (int j = 0; j < 8; ++j)
        h2[(mb + j) * 64 + col] = (_Float16)fmaxf(c[j] + bias, 0.0f);
    }
    lds_fence();

    // layer 3: 16x64 @ 64x32 -> pf (16x32), scatter atomic-max into vox
    v16h a20 = load_a_f16(h2, 64, 0);
    v16h a21 = load_a_f16(h2, 64, 32);
#pragma unroll
    for (int t = 0; t < 2; ++t) {
      v8f c = {};
      c = wmma32(a20, b3f[t][0], c);
      c = wmma32(a21, b3f[t][1], c);
      const int col = t * 16 + n;
      const float bias = s_b3[col];
#pragma unroll
      for (int j = 0; j < 8; ++j) {
        const int prow = row0 + mb + j;
        if (prow < N) {
          float v = fmaxf(c[j] + bias, 0.0f);
          pf16[(size_t)prow * 32 + col] = (_Float16)v;
          atomicMax((int*)&vox[(size_t)pv[prow] * 32 + col], __float_as_int(v));
        }
      }
    }
  }
}

// ===========================================================================
// Kernel 2: voxel MLP: v1 = relu(bn(vox@ve_w1)), vf = relu(bn(v1@ve_w2)),
//           aux = relu(bn(v1@aux_w1)) @ aux_w2  (V x 20)
// ===========================================================================
__global__ __launch_bounds__(256) void k_voxel_encode(
    const float* __restrict__ vox,
    const float* __restrict__ vw1, const float* __restrict__ vg1, const float* __restrict__ vb1,
    const float* __restrict__ vw2, const float* __restrict__ vg2, const float* __restrict__ vb2,
    const float* __restrict__ aw1, const float* __restrict__ ag1, const float* __restrict__ ab1,
    const float* __restrict__ aw2,
    _Float16* __restrict__ vf16, float* __restrict__ aux, int V) {
  __shared__ alignas(16) _Float16 s_v1[32 * 32]; __shared__ float s_bv1[32];
  __shared__ alignas(16) _Float16 s_v2[32 * 32]; __shared__ float s_bv2[32];
  __shared__ alignas(16) _Float16 s_a1[32 * 32]; __shared__ float s_ba1[32];
  __shared__ alignas(16) _Float16 s_a2[32 * 32]; // aux_w2 zero-pad 32x20->32x32
  __shared__ alignas(16) _Float16 s_t[8][16 * 32];

  const int tid = threadIdx.x;
  const float inv = rsqrtf(1.0f + 1e-5f);
  for (int i = tid; i < 32 * 32; i += 256) {
    int c = i & 31, rw = i >> 5;
    s_v1[i] = (_Float16)(vw1[i] * (vg1[c] * inv));
    s_v2[i] = (_Float16)(vw2[i] * (vg2[c] * inv));
    s_a1[i] = (_Float16)(aw1[i] * (ag1[c] * inv));
    s_a2[i] = (c < NCLS) ? (_Float16)aw2[rw * NCLS + c] : (_Float16)0.0f;
  }
  for (int i = tid; i < 32; i += 256) { s_bv1[i] = vb1[i]; s_bv2[i] = vb2[i]; s_ba1[i] = ab1[i]; }
  __syncthreads();

  const int wave = tid >> 5, lane = tid & 31;
  const int n = lane & 15, mb = (lane >> 4) * 8, kb = (lane >> 4) * 8;

  v16h bv1[2], bv2[2], ba1[2], ba2[2];
#pragma unroll
  for (int t = 0; t < 2; ++t) {
    bv1[t] = load_b_f16(s_v1, 32, 0, t * 16);
    bv2[t] = load_b_f16(s_v2, 32, 0, t * 16);
    ba1[t] = load_b_f16(s_a1, 32, 0, t * 16);
    ba2[t] = load_b_f16(s_a2, 32, 0, t * 16);
  }

  const int ntiles = (V + 15) >> 4;
  const int nwaves = gridDim.x * 8;
  for (int tile = blockIdx.x * 8 + wave; tile < ntiles; tile += nwaves) {
    const int row0 = tile << 4;
    const int myrow = row0 + n;
    const int r = myrow < V ? myrow : V - 1;

    // A-frag of vox row: vectorized f32 loads (rows are 128B aligned);
    // empty voxels are 0 (memset), matching reference isfinite->0 since pf>=0.
    const float* vr = vox + (size_t)r * 32;
    v4f f0 = *(const v4f*)(vr + kb);
    v4f f1 = *(const v4f*)(vr + kb + 4);
    v4f f2 = *(const v4f*)(vr + 16 + kb);
    v4f f3 = *(const v4f*)(vr + 16 + kb + 4);
    v16h a0;
#pragma unroll
    for (int i = 0; i < 4; ++i) {
      a0[i]      = (_Float16)f0[i];
      a0[4 + i]  = (_Float16)f1[i];
      a0[8 + i]  = (_Float16)f2[i];
      a0[12 + i] = (_Float16)f3[i];
    }

    _Float16* tl = s_t[wave];
    // v1
#pragma unroll
    for (int t = 0; t < 2; ++t) {
      v8f c = {};
      c = wmma32(a0, bv1[t], c);
      const int col = t * 16 + n; const float bias = s_bv1[col];
#pragma unroll
      for (int j = 0; j < 8; ++j)
        tl[(mb + j) * 32 + col] = (_Float16)fmaxf(c[j] + bias, 0.0f);
    }
    lds_fence();
    v16h av1 = load_a_f16(tl, 32, 0);

    // vf (stored for gather in fuse stage)
#pragma unroll
    for (int t = 0; t < 2; ++t) {
      v8f c = {};
      c = wmma32(av1, bv2[t], c);
      const int col = t * 16 + n; const float bias = s_bv2[col];
#pragma unroll
      for (int j = 0; j < 8; ++j) {
        const int vrow = row0 + mb + j;
        if (vrow < V) vf16[(size_t)vrow * 32 + col] = (_Float16)fmaxf(c[j] + bias, 0.0f);
      }
    }

    // aux hidden (reuse wave tile), then aux head
#pragma unroll
    for (int t = 0; t < 2; ++t) {
      v8f c = {};
      c = wmma32(av1, ba1[t], c);
      const int col = t * 16 + n; const float bias = s_ba1[col];
#pragma unroll
      for (int j = 0; j < 8; ++j)
        tl[(mb + j) * 32 + col] = (_Float16)fmaxf(c[j] + bias, 0.0f);
    }
    lds_fence();
    v16h ah = load_a_f16(tl, 32, 0);
#pragma unroll
    for (int t = 0; t < 2; ++t) {
      v8f c = {};
      c = wmma32(ah, ba2[t], c);
      const int col = t * 16 + n;
      if (col < NCLS) {
#pragma unroll
        for (int j = 0; j < 8; ++j) {
          const int vrow = row0 + mb + j;
          if (vrow < V) aux[(size_t)vrow * NCLS + col] = c[j];
        }
      }
    }
  }
}

// ===========================================================================
// Kernel 3: fused = relu(bn([vf[pv] | pf] @ fu_w)); store f16; block-reduced
//           batch segment sums + counts.
// ===========================================================================
__global__ __launch_bounds__(256) void k_fuse(
    const float* __restrict__ points, const int* __restrict__ pv,
    const _Float16* __restrict__ pf16, const _Float16* __restrict__ vf16,
    const float* __restrict__ fw, const float* __restrict__ fg, const float* __restrict__ fb,
    _Float16* __restrict__ fused16, float* __restrict__ bsums, float* __restrict__ bcnt, int N) {
  __shared__ alignas(16) _Float16 s_w[64 * 64];
  __shared__ float s_b[64];
  __shared__ float s_acc[4 * 64];
  __shared__ float s_cnt[4];

  const int tid = threadIdx.x;
  const float inv = rsqrtf(1.0f + 1e-5f);
  for (int i = tid; i < 64 * 64; i += 256) s_w[i] = (_Float16)(fw[i] * (fg[i & 63] * inv));
  for (int i = tid; i < 64; i += 256) s_b[i] = fb[i];
  for (int i = tid; i < 256; i += 256) s_acc[i] = 0.0f;
  if (tid < 4) s_cnt[tid] = 0.0f;
  __syncthreads();

  const int wave = tid >> 5, lane = tid & 31;
  const int n = lane & 15, mb = (lane >> 4) * 8, kb = (lane >> 4) * 8;

  v16h bf[4][2];
#pragma unroll
  for (int t = 0; t < 4; ++t) {
    bf[t][0] = load_b_f16(s_w, 64, 0,  t * 16);
    bf[t][1] = load_b_f16(s_w, 64, 32, t * 16);
  }

  const int ntiles = (N + 15) >> 4;
  const int nwaves = gridDim.x * 8;
  for (int tile = blockIdx.x * 8 + wave; tile < ntiles; tile += nwaves) {
    const int row0 = tile << 4;
    const int myrow = row0 + n;
    const int r = myrow < N ? myrow : N - 1;

    // A = [ vf[pv[r]] (K 0..31) | pf[r] (K 32..63) ], b128 gather loads
    const _Float16* gv = vf16 + (size_t)pv[r] * 32;
    const _Float16* gp = pf16 + (size_t)r * 32;
    v16h alo = CAT16(*(const v8h*)(gv + kb), *(const v8h*)(gv + 16 + kb));
    v16h ahi = CAT16(*(const v8h*)(gp + kb), *(const v8h*)(gp + 16 + kb));

    // one point per tile-row counted once (lanes 0-15)
    if (lane < 16 && myrow < N)
      atomicAdd(&s_cnt[(int)points[(size_t)myrow * 5]], 1.0f);

#pragma unroll
    for (int t = 0; t < 4; ++t) {
      v8f c = {};
      c = wmma32(alo, bf[t][0], c);
      c = wmma32(ahi, bf[t][1], c);
      const int col = t * 16 + n; const float bias = s_b[col];
#pragma unroll
      for (int j = 0; j < 8; ++j) {
        const int prow = row0 + mb + j;
        if (prow < N) {
          float v = fmaxf(c[j] + bias, 0.0f);
          fused16[(size_t)prow * 64 + col] = (_Float16)v;
          atomicAdd(&s_acc[(int)points[(size_t)prow * 5] * 64 + col], v);
        }
      }
    }
  }

  __syncthreads();
  for (int i = tid; i < 256; i += 256) atomicAdd(&bsums[i], s_acc[i]);
  if (tid < 4) atomicAdd(&bcnt[tid], s_cnt[tid]);
}

// ===========================================================================
// Kernel 4: SE gate (B=4), one block.
// ===========================================================================
__global__ __launch_bounds__(256) void k_gate(
    const float* __restrict__ bsums, const float* __restrict__ bcnt,
    const float* __restrict__ sw1, const float* __restrict__ sb1,
    const float* __restrict__ sw2, const float* __restrict__ sb2,
    float* __restrict__ gate) {
  __shared__ float s_mean[4 * 64];
  __shared__ float s_h[4 * 4];
  const int tid = threadIdx.x;
  if (tid < 256) s_mean[tid] = bsums[tid] / fmaxf(bcnt[tid >> 6], 1.0f);
  __syncthreads();
  if (tid < 16) {
    const int b = tid >> 2, j = tid & 3;
    float h = sb1[j];
    for (int c = 0; c < 64; ++c) h = fmaf(s_mean[b * 64 + c], sw1[c * 4 + j], h);
    s_h[tid] = fmaxf(h, 0.0f);
  }
  __syncthreads();
  if (tid < 256) {
    const int b = tid >> 6, c = tid & 63;
    float g = sb2[c];
    for (int j = 0; j < 4; ++j) g = fmaf(s_h[b * 4 + j], sw2[j * 64 + c], g);
    gate[tid] = 1.0f / (1.0f + expf(-g));
  }
}

// ===========================================================================
// Kernel 5: out = relu(bn((fused*(1+gate[bid])) @ cls_w1)) @ cls_w2  (N x 20)
// ===========================================================================
__global__ __launch_bounds__(256) void k_classify(
    const float* __restrict__ points, const _Float16* __restrict__ fused16,
    const float* __restrict__ gate,
    const float* __restrict__ cw1, const float* __restrict__ cg1, const float* __restrict__ cb1,
    const float* __restrict__ cw2,
    float* __restrict__ out, int N) {
  __shared__ alignas(16) _Float16 s_w1[64 * 32]; __shared__ float s_b1[32];
  __shared__ alignas(16) _Float16 s_w2[32 * 32]; // cls_w2 zero-pad 32x20->32x32
  __shared__ alignas(16) _Float16 s_g16[4 * 64]; // 1 + gate, f16
  __shared__ alignas(16) _Float16 s_t[8][16 * 32];

  const int tid = threadIdx.x;
  const float inv = rsqrtf(1.0f + 1e-5f);
  for (int i = tid; i < 64 * 32; i += 256) s_w1[i] = (_Float16)(cw1[i] * (cg1[i & 31] * inv));
  for (int i = tid; i < 32 * 32; i += 256) {
    int c = i & 31, rw = i >> 5;
    s_w2[i] = (c < NCLS) ? (_Float16)cw2[rw * NCLS + c] : (_Float16)0.0f;
  }
  for (int i = tid; i < 32; i += 256) s_b1[i] = cb1[i];
  for (int i = tid; i < 256; i += 256) s_g16[i] = (_Float16)(1.0f + gate[i]);
  __syncthreads();

  const int wave = tid >> 5, lane = tid & 31;
  const int n = lane & 15, mb = (lane >> 4) * 8, kb = (lane >> 4) * 8;

  v16h b1f[2][2], b2f[2];
#pragma unroll
  for (int t = 0; t < 2; ++t) {
    b1f[t][0] = load_b_f16(s_w1, 32, 0,  t * 16);
    b1f[t][1] = load_b_f16(s_w1, 32, 32, t * 16);
    b2f[t]    = load_b_f16(s_w2, 32, 0,  t * 16);
  }

  const int ntiles = (N + 15) >> 4;
  const int nwaves = gridDim.x * 8;
  for (int tile = blockIdx.x * 8 + wave; tile < ntiles; tile += nwaves) {
    const int row0 = tile << 4;
    const int myrow = row0 + n;
    const int r = myrow < N ? myrow : N - 1;
    const int bid = (int)points[(size_t)r * 5];
    const _Float16* gb = &s_g16[bid * 64];
    const _Float16* fr = fused16 + (size_t)r * 64;

    // b128 loads of fused row; packed-half multiply with (1+gate) frags
    v8h f0 = *(const v8h*)(fr + kb)      * *(const v8h*)(gb + kb);
    v8h f1 = *(const v8h*)(fr + 16 + kb) * *(const v8h*)(gb + 16 + kb);
    v8h f2 = *(const v8h*)(fr + 32 + kb) * *(const v8h*)(gb + 32 + kb);
    v8h f3 = *(const v8h*)(fr + 48 + kb) * *(const v8h*)(gb + 48 + kb);
    v16h alo = CAT16(f0, f1);
    v16h ahi = CAT16(f2, f3);

    _Float16* tl = s_t[wave];
#pragma unroll
    for (int t = 0; t < 2; ++t) {
      v8f c = {};
      c = wmma32(alo, b1f[t][0], c);
      c = wmma32(ahi, b1f[t][1], c);
      const int col = t * 16 + n; const float bias = s_b1[col];
#pragma unroll
      for (int j = 0; j < 8; ++j)
        tl[(mb + j) * 32 + col] = (_Float16)fmaxf(c[j] + bias, 0.0f);
    }
    lds_fence();
    v16h ah = load_a_f16(tl, 32, 0);
#pragma unroll
    for (int t = 0; t < 2; ++t) {
      v8f c = {};
      c = wmma32(ah, b2f[t], c);
      const int col = t * 16 + n;
      if (col < NCLS) {
#pragma unroll
        for (int j = 0; j < 8; ++j) {
          const int prow = row0 + mb + j;
          if (prow < N) out[(size_t)prow * NCLS + col] = c[j];
        }
      }
    }
  }
}

// ===========================================================================
extern "C" void kernel_launch(void* const* d_in, const int* in_sizes, int n_in,
                              void* d_out, int out_size, void* d_ws, size_t ws_size,
                              hipStream_t stream) {
  const float* points = (const float*)d_in[0];
  const int*   pv     = (const int*)d_in[1];
  const float *pe_w1 = (const float*)d_in[2],  *pe_g1 = (const float*)d_in[3],  *pe_b1 = (const float*)d_in[4];
  const float *pe_w2 = (const float*)d_in[5],  *pe_g2 = (const float*)d_in[6],  *pe_b2 = (const float*)d_in[7];
  const float *pe_w3 = (const float*)d_in[8],  *pe_g3 = (const float*)d_in[9],  *pe_b3 = (const float*)d_in[10];
  const float *ve_w1 = (const float*)d_in[11], *ve_g1 = (const float*)d_in[12], *ve_b1 = (const float*)d_in[13];
  const float *ve_w2 = (const float*)d_in[14], *ve_g2 = (const float*)d_in[15], *ve_b2 = (const float*)d_in[16];
  const float *fu_w  = (const float*)d_in[17], *fu_g  = (const float*)d_in[18], *fu_b  = (const float*)d_in[19];
  const float *se_w1 = (const float*)d_in[20], *se_b1 = (const float*)d_in[21];
  const float *se_w2 = (const float*)d_in[22], *se_b2 = (const float*)d_in[23];
  const float *cl_w1 = (const float*)d_in[24], *cl_g1 = (const float*)d_in[25], *cl_b1 = (const float*)d_in[26];
  const float *cl_w2 = (const float*)d_in[27];
  const float *ax_w1 = (const float*)d_in[28], *ax_g1 = (const float*)d_in[29], *ax_b1 = (const float*)d_in[30];
  const float *ax_w2 = (const float*)d_in[31];

  const int N = in_sizes[0] / 5;
  const int V = VOX;

  // workspace layout
  char* ws = (char*)d_ws;
  size_t off = 0;
  auto take = [&](size_t bytes) { char* p = ws + off; off += (bytes + 255) & ~(size_t)255; return p; };
  _Float16* pf16    = (_Float16*)take((size_t)N * 32 * sizeof(_Float16));
  float*    vox     = (float*)   take((size_t)V * 32 * sizeof(float));
  _Float16* vf16    = (_Float16*)take((size_t)V * 32 * sizeof(_Float16));
  _Float16* fused16 = (_Float16*)take((size_t)N * 64 * sizeof(_Float16));
  float*    stats   = (float*)   take((4 * 64 + 4 + 4 * 64) * sizeof(float));
  float* bsums = stats;            // 256
  float* bcnt  = stats + 256;      // 4
  float* gate  = stats + 260;      // 256

  float* out = (float*)d_out;
  float* aux = out + (size_t)N * NCLS;

  // per-call zeroing (graph-capture safe)
  hipMemsetAsync(vox, 0, (size_t)V * 32 * sizeof(float), stream);
  hipMemsetAsync(stats, 0, 260 * sizeof(float), stream);

  const int tilesN = (N + 15) / 16;
  const int tilesV = (V + 15) / 16;
  const int blkN = (tilesN + 7) / 8;
  const int blkV = (tilesV + 7) / 8;

  k_point_encode<<<blkN, 256, 0, stream>>>(points, pv,
      pe_w1, pe_g1, pe_b1, pe_w2, pe_g2, pe_b2, pe_w3, pe_g3, pe_b3,
      pf16, vox, N);
  k_voxel_encode<<<blkV, 256, 0, stream>>>(vox,
      ve_w1, ve_g1, ve_b1, ve_w2, ve_g2, ve_b2,
      ax_w1, ax_g1, ax_b1, ax_w2, vf16, aux, V);
  k_fuse<<<blkN, 256, 0, stream>>>(points, pv, pf16, vf16,
      fu_w, fu_g, fu_b, fused16, bsums, bcnt, N);
  k_gate<<<1, 256, 0, stream>>>(bsums, bcnt, se_w1, se_b1, se_w2, se_b2, gate);
  k_classify<<<blkN, 256, 0, stream>>>(points, fused16, gate,
      cl_w1, cl_g1, cl_b1, cl_w2, out, N);
}
